// GRU1_71485435674721
// MI455X (gfx1250) — matrix operand order repaired and checked by reading, not compiled, not run on hardware
//
#include <hip/hip_runtime.h>
#include <hip/hip_bf16.h>
#include <cstdint>

// Problem dims (match reference)
#define T_LEN 512
#define BATCH 128
#define DIN   512
#define DH    512

typedef __attribute__((ext_vector_type(16))) __bf16 bf16x16;
typedef __attribute__((ext_vector_type(8)))  float  f32x8;
typedef __attribute__((ext_vector_type(4)))  unsigned int u32x4;
typedef __attribute__((ext_vector_type(8)))  int    i32x8;
typedef __attribute__((ext_vector_type(4)))  int    i32x4;

#if defined(__has_builtin)
#if __has_builtin(__builtin_amdgcn_tensor_load_to_lds) && \
    __has_builtin(__builtin_amdgcn_s_wait_tensorcnt)
#define GRU_HAVE_TDM 1
#endif
#endif

union FragU {
    bf16x16 v;
    uint4   q[2];
};

__device__ __forceinline__ unsigned short f32_to_bf16_rne(float f) {
    union { float f; uint32_t u; } c;
    c.f = f;
    uint32_t u = c.u;
    uint32_t r = u + 0x7FFFu + ((u >> 16) & 1u);
    return (unsigned short)(r >> 16);
}

__device__ __forceinline__ float sigmoidf_dev(float x) {
    return 1.0f / (1.0f + __expf(-x));
}

// LDS byte-offset of a __shared__ object (generic -> AS(3) -> int)
__device__ __forceinline__ uint32_t lds_addr_of(void* p) {
    return (uint32_t)(uintptr_t)(__attribute__((address_space(3))) void*)p;
}

#if defined(GRU_HAVE_TDM)
// ---------------------------------------------------------------------------
// TDM: DMA a 2D bf16 tile [tile_rows x 512] (row stride 512 elements) from
// global memory into LDS. Issued once per block by a single wave; caller must
// s_wait_tensorcnt + barrier before consuming.
// D# layout per CDNA5 ISA ch.8 (group0: count/lds/global/type;
// group1: data_size=2B, tensor_dim0=512, tensor_dim1=tensor_rows,
// tile_dim0=512, tile_dim1=tile_rows, tensor_dim0_stride=512).
// ---------------------------------------------------------------------------
__device__ __forceinline__ void tdm_load_tile_bf16(const unsigned short* gsrc,
                                                   void* lds_dst,
                                                   int tile_rows,
                                                   int tensor_rows) {
    const uint64_t ga = (uint64_t)(uintptr_t)gsrc;
    const uint32_t lo = lds_addr_of(lds_dst);
    u32x4 g0 = {1u, lo, (uint32_t)ga,
                (((uint32_t)(ga >> 32)) & 0x01FFFFFFu) | 0x80000000u};
    i32x8 g1 = {0x00010000,                          // data_size = 1 (2 bytes)
                (int)(512u << 16),                   // tensor_dim0 = 512
                (int)(((uint32_t)tensor_rows) << 16),// tensor_dim1
                (int)(512u << 16),                   // tile_dim0 = 512
                tile_rows,                           // tile_dim1 (tile_dim2 = 0)
                512,                                 // tensor_dim0_stride = 512
                0, 0};
    i32x4 z4 = {0, 0, 0, 0};
#if defined(__clang_major__) && (__clang_major__ >= 23)
    i32x8 z8 = {0, 0, 0, 0, 0, 0, 0, 0};
    __builtin_amdgcn_tensor_load_to_lds(g0, g1, z4, z4, z8, 0);
#else
    __builtin_amdgcn_tensor_load_to_lds(g0, g1, z4, z4, 0);
#endif
}
#endif

// ---------------------------------------------------------------------------
// f32 -> bf16 conversion (seq + weights)
// ---------------------------------------------------------------------------
__global__ void gru_cvt_bf16(const float* __restrict__ src,
                             unsigned short* __restrict__ dst, int n) {
    int i = blockIdx.x * blockDim.x + threadIdx.x;
    if (i < n) dst[i] = f32_to_bf16_rne(src[i]);
}

__global__ void gru_zero_u16(unsigned short* __restrict__ p, int n) {
    int i = blockIdx.x * blockDim.x + threadIdx.x;
    if (i < n) p[i] = 0;
}

// ---------------------------------------------------------------------------
// Input projections: out[m,n] = sum_k seq[m,k] * W[n,k] + bias[n]
// A: [M=T*B, K=DIN] bf16 row-major, W: [DH, DIN] bf16 row-major, out f32 [M, DH]
// grid = (DH/64, M/128, 2), block = 256 (8 waves).
// The 64-row W panel (64 KB) shared by all 8 waves is brought in with one TDM
// tensor_load_to_lds; A-stream is double-buffered from global.
// ---------------------------------------------------------------------------
__global__ __launch_bounds__(256)
void gru_input_proj(const unsigned short* __restrict__ seq_bf,
                    const unsigned short* __restrict__ Wz_bf,
                    const unsigned short* __restrict__ Wn_bf,
                    const float* __restrict__ bz,
                    const float* __restrict__ bn,
                    float* __restrict__ out_z,
                    float* __restrict__ out_n) {
    __shared__ unsigned short ldsW[64 * DIN];  // 64 KB

    const int tid  = threadIdx.x;
    const int lane = tid & 31;
    const int wave = tid >> 5;
    const int m0   = blockIdx.y * 128 + wave * 16;
    const int n0   = blockIdx.x * 64;

    const unsigned short* __restrict__ W    = blockIdx.z ? Wn_bf : Wz_bf;
    const float* __restrict__          bias = blockIdx.z ? bn    : bz;
    float* __restrict__                out  = blockIdx.z ? out_n : out_z;

#if defined(GRU_HAVE_TDM)
    if (wave == 0) {
        tdm_load_tile_bf16(W + (size_t)n0 * DIN, ldsW, /*tile_rows=*/64,
                           /*tensor_rows=*/DH);
        __builtin_amdgcn_s_wait_tensorcnt(0);
    }
    __syncthreads();
#else
    // Fallback: cooperative fill, 64 rows x 512 cols bf16 = 4096 uint4.
    for (int i = tid; i < 64 * (DIN / 8); i += 256) {
        const int row = i >> 6;   // 64 uint4 per row
        const int c8  = i & 63;
        *(uint4*)&ldsW[row * DIN + c8 * 8] =
            *(const uint4*)(W + (size_t)(n0 + row) * DIN + c8 * 8);
    }
    __syncthreads();
#endif

    const int mrow = lane & 15;
    const int hs   = lane >> 4;

    const f32x8 zero8 = {0.f, 0.f, 0.f, 0.f, 0.f, 0.f, 0.f, 0.f};
    f32x8 acc[4];
#pragma unroll
    for (int j = 0; j < 4; ++j) acc[j] = zero8;

    const unsigned short* __restrict__ Arow =
        seq_bf + (size_t)(m0 + mrow) * DIN + 16 * hs;

    // Double-buffered A stream
    uint4 a0 = *(const uint4*)(Arow);
    uint4 a1 = *(const uint4*)(Arow + 8);

    for (int k0 = 0; k0 < DIN; k0 += 32) {
        FragU a;
        a.q[0] = a0;
        a.q[1] = a1;
        if (k0 + 32 < DIN) {
            a0 = *(const uint4*)(Arow + k0 + 32);
            a1 = *(const uint4*)(Arow + k0 + 40);
        }
#pragma unroll
        for (int j = 0; j < 4; ++j) {
            FragU b;
            const unsigned short* bp = &ldsW[(j * 16 + mrow) * DIN + k0 + 16 * hs];
            b.q[0] = *(const uint4*)(bp);
            b.q[1] = *(const uint4*)(bp + 8);
            acc[j] = __builtin_amdgcn_wmma_f32_16x16x32_bf16(
                false, a.v, false, b.v, (short)0, acc[j], false, false);
        }
    }

#pragma unroll
    for (int j = 0; j < 4; ++j) {
        const int   n  = n0 + j * 16 + mrow;
        const float bv = bias[n];
#pragma unroll
        for (int v = 0; v < 8; ++v) {
            const int m = m0 + 8 * hs + v;
            out[(size_t)m * DH + n] = acc[j][v] + bv;
        }
    }
}

// ---------------------------------------------------------------------------
// One GRU timestep.
// grid = (DH/128, BATCH/16), block = 256 (8 waves).
// All 8 waves share the 16-row h_prev strip -> one TDM tensor_load_to_lds
// (2D tile: 512 x 16 of 2-byte elements) fills LDS; waves then feed A
// fragments with ds_load_b128. W fragments stream from L2, double-buffered.
// ---------------------------------------------------------------------------
__global__ __launch_bounds__(256)
void gru_step(const unsigned short* __restrict__ h_bf_prev,  // [BATCH, DH] bf16
              const unsigned short* __restrict__ Whz_bf,     // [DH, DH] bf16
              const unsigned short* __restrict__ Whn_bf,     // [DH, DH] bf16
              const float* __restrict__ bhz,
              const float* __restrict__ bhn,
              const float* __restrict__ iz_t,                // [BATCH, DH] f32
              const float* __restrict__ inn_t,               // [BATCH, DH] f32
              const float* __restrict__ h_prev_f32,          // [BATCH, DH] f32
              float* __restrict__ h_out_f32,                 // [BATCH, DH] f32
              unsigned short* __restrict__ h_bf_out,         // [BATCH, DH] bf16
              int is_first) {
    __shared__ unsigned short lds_h[16 * DH];  // 16 KB

    const int tid  = threadIdx.x;
    const int lane = tid & 31;
    const int wave = tid >> 5;
    const int n0   = blockIdx.x * 128 + wave * 16;
    const int m0   = blockIdx.y * 16;

#if defined(GRU_HAVE_TDM)
    if (wave == 0) {
        tdm_load_tile_bf16(h_bf_prev + (size_t)m0 * DH, lds_h, /*tile_rows=*/16,
                           /*tensor_rows=*/BATCH);
        __builtin_amdgcn_s_wait_tensorcnt(0);
    }
    __syncthreads();
#else
    // Fallback: cooperative copy (16 rows x 512 cols bf16 = 1024 uint4)
    for (int i = tid; i < 16 * (DH / 8); i += 256) {
        const int row = i >> 6;
        const int c8  = i & 63;
        *(uint4*)&lds_h[row * DH + c8 * 8] =
            *(const uint4*)(h_bf_prev + (size_t)(m0 + row) * DH + c8 * 8);
    }
    __syncthreads();
#endif

    const int mrow = lane & 15;
    const int hs   = lane >> 4;

    const f32x8 zero8 = {0.f, 0.f, 0.f, 0.f, 0.f, 0.f, 0.f, 0.f};
    f32x8 acc_z = zero8;
    f32x8 acc_n = zero8;

    const unsigned short* __restrict__ Az  = &lds_h[mrow * DH + 16 * hs];
    const unsigned short* __restrict__ Bzp = Whz_bf + (size_t)(n0 + mrow) * DH + 16 * hs;
    const unsigned short* __restrict__ Bnp = Whn_bf + (size_t)(n0 + mrow) * DH + 16 * hs;

    // Double-buffered weight streams
    uint4 bz0 = *(const uint4*)(Bzp);
    uint4 bz1 = *(const uint4*)(Bzp + 8);
    uint4 bn0 = *(const uint4*)(Bnp);
    uint4 bn1 = *(const uint4*)(Bnp + 8);

    for (int k0 = 0; k0 < DH; k0 += 32) {
        FragU a, bzf, bnf;
        a.q[0]   = *(const uint4*)(Az + k0);       // ds_load_b128
        a.q[1]   = *(const uint4*)(Az + k0 + 8);
        bzf.q[0] = bz0;
        bzf.q[1] = bz1;
        bnf.q[0] = bn0;
        bnf.q[1] = bn1;
        if (k0 + 32 < DH) {
            bz0 = *(const uint4*)(Bzp + k0 + 32);
            bz1 = *(const uint4*)(Bzp + k0 + 48);
            bn0 = *(const uint4*)(Bnp + k0 + 32);
            bn1 = *(const uint4*)(Bnp + k0 + 48);
        }
        acc_z = __builtin_amdgcn_wmma_f32_16x16x32_bf16(
            false, a.v, false, bzf.v, (short)0, acc_z, false, false);
        acc_n = __builtin_amdgcn_wmma_f32_16x16x32_bf16(
            false, a.v, false, bnf.v, (short)0, acc_n, false, false);
    }

    const int   n   = n0 + mrow;
    const float bzv = bhz[n];
    const float bnv = bhn[n];

#pragma unroll
    for (int v = 0; v < 8; ++v) {
        const int    m   = m0 + 8 * hs + v;
        const size_t idx = (size_t)m * DH + n;
        const float  hp  = is_first ? 0.0f : h_prev_f32[idx];
        const float  z   = sigmoidf_dev(iz_t[idx] + acc_z[v] + bzv);
        const float  nn  = tanhf(inn_t[idx] + acc_n[v] + bnv);
        const float  h   = (1.0f - z) * nn + z * hp;
        h_out_f32[idx] = h;
        h_bf_out[idx]  = f32_to_bf16_rne(h);
    }
}

// ---------------------------------------------------------------------------
// Host launcher
// ---------------------------------------------------------------------------
extern "C" void kernel_launch(void* const* d_in, const int* in_sizes, int n_in,
                              void* d_out, int out_size, void* d_ws, size_t ws_size,
                              hipStream_t stream) {
    const float* seq  = (const float*)d_in[0];
    const float* W_iz = (const float*)d_in[1];
    const float* b_iz = (const float*)d_in[2];
    const float* W_in = (const float*)d_in[3];
    const float* b_in = (const float*)d_in[4];
    const float* W_hz = (const float*)d_in[5];
    const float* b_hz = (const float*)d_in[6];
    const float* W_hn = (const float*)d_in[7];
    const float* b_hn = (const float*)d_in[8];
    float* H = (float*)d_out;  // [1, T, B, DH]

    // Workspace layout
    char*  ws  = (char*)d_ws;
    size_t off = 0;
    unsigned short* seq_bf = (unsigned short*)(ws + off); off += (size_t)T_LEN * BATCH * DIN * 2;
    unsigned short* Wiz_bf = (unsigned short*)(ws + off); off += (size_t)DH * DIN * 2;
    unsigned short* Win_bf = (unsigned short*)(ws + off); off += (size_t)DH * DIN * 2;
    unsigned short* Whz_bf = (unsigned short*)(ws + off); off += (size_t)DH * DH * 2;
    unsigned short* Whn_bf = (unsigned short*)(ws + off); off += (size_t)DH * DH * 2;
    float* iz  = (float*)(ws + off); off += (size_t)T_LEN * BATCH * DH * 4;
    float* inn = (float*)(ws + off); off += (size_t)T_LEN * BATCH * DH * 4;
    unsigned short* hbf0 = (unsigned short*)(ws + off); off += (size_t)BATCH * DH * 2;
    unsigned short* hbf1 = (unsigned short*)(ws + off); off += (size_t)BATCH * DH * 2;

    // --- Phase 0: precision conversion ---
    {
        const int n_seq = T_LEN * BATCH * DIN;
        gru_cvt_bf16<<<(n_seq + 255) / 256, 256, 0, stream>>>(seq, seq_bf, n_seq);
        const int n_w = DH * DIN;
        gru_cvt_bf16<<<(n_w + 255) / 256, 256, 0, stream>>>(W_iz, Wiz_bf, n_w);
        gru_cvt_bf16<<<(n_w + 255) / 256, 256, 0, stream>>>(W_in, Win_bf, n_w);
        gru_cvt_bf16<<<(n_w + 255) / 256, 256, 0, stream>>>(W_hz, Whz_bf, n_w);
        gru_cvt_bf16<<<(n_w + 255) / 256, 256, 0, stream>>>(W_hn, Whn_bf, n_w);
        const int n_h = BATCH * DH;
        gru_zero_u16<<<(n_h + 255) / 256, 256, 0, stream>>>(hbf1, n_h);
    }

    // --- Phase 1: input projections for all timesteps (big WMMA GEMM) ---
    {
        dim3 grid(DH / 64, (T_LEN * BATCH) / 128, 2);
        gru_input_proj<<<grid, 256, 0, stream>>>(seq_bf, Wiz_bf, Win_bf,
                                                 b_iz, b_in, iz, inn);
    }

    // --- Phase 2: sequential recurrence, one small WMMA kernel per step ---
    for (int t = 0; t < T_LEN; ++t) {
        unsigned short* hin  = (t & 1) ? hbf0 : hbf1;  // even t reads hbf1
        unsigned short* hout = (t & 1) ? hbf1 : hbf0;
        const float* hprev_f =
            (t == 0) ? H : (H + (size_t)(t - 1) * BATCH * DH);
        dim3 grid(DH / 128, BATCH / 16);
        gru_step<<<grid, 256, 0, stream>>>(
            hin, Whz_bf, Whn_bf, b_hz, b_hn,
            iz + (size_t)t * BATCH * DH, inn + (size_t)t * BATCH * DH,
            hprev_f, H + (size_t)t * BATCH * DH, hout, t == 0 ? 1 : 0);
    }
}
